// Downsampler_80779744903457
// MI455X (gfx1250) — matrix operand order repaired and checked by reading, not compiled, not run on hardware
//
#include <hip/hip_runtime.h>
#include <hip/hip_bf16.h>
#include <stdint.h>

typedef __attribute__((ext_vector_type(16))) __bf16 v16bf;
typedef __attribute__((ext_vector_type(8)))  float  v8f;
typedef __attribute__((ext_vector_type(4)))  unsigned tdm_g0_t;
typedef __attribute__((ext_vector_type(8)))  int      tdm_g1_t;
typedef __attribute__((ext_vector_type(4)))  int      tdm_g24_t;

constexpr int IH = 256, IW = 256, NIMG = 8;
constexpr int CIN = 34, COUT = 64;
constexpr int KTOT = 306, KPAD = 320;     // 34*9 padded to multiple of 32
constexpr int PIX  = 64;                   // pixels per conv workgroup
constexpr int KSTR = 328;                  // K-stride: 41*16B rows -> 16B-aligned, conflict-free b128
constexpr int RW   = 68;                   // raw-tile row stride (floats); cols 0..65 used
constexpr float TAU = 0.1f;
constexpr float BN_EPS = 1e-5f;
// fused-TV tiling
constexpr int TVT = 48, HALO = 10, LOADT = TVT + 2 * HALO /*68*/, LSTR = LOADT + 1 /*69*/;

#if defined(__has_builtin)
# if __has_builtin(__builtin_amdgcn_tensor_load_to_lds)
#  define HAVE_TDM 1
# endif
#endif
#ifndef HAVE_TDM
# define HAVE_TDM 0
#endif

__device__ __forceinline__ unsigned short f32_bf16(float f) {
  union { float f; unsigned u; } x; x.f = f;
  unsigned u = x.u;
  unsigned r = u + 0x7FFFu + ((u >> 16) & 1u);   // round-to-nearest-even
  return (unsigned short)(r >> 16);
}

// LDS byte offset of a generic pointer to a __shared__ object.
__device__ __forceinline__ unsigned lds_off(const void* p) {
  return (unsigned)(size_t)p;
}

// ASYNCcnt-tracked global->LDS copy (per-lane addresses, honors EXEC).
__device__ __forceinline__ void async_ld_b32(unsigned ldsoff, const void* gptr) {
  asm volatile("global_load_async_to_lds_b32 %0, %1, off"
               :: "v"(ldsoff), "v"(gptr) : "memory");
}
__device__ __forceinline__ void wait_async0() {
  asm volatile("s_wait_asynccnt 0x0" ::: "memory");
}

// ---- K0: pack conv_w [64][34][3][3] f32 -> Wt [64][328] bf16 (K = c*9+ky*3+kx, zero-padded)
__global__ void wt_xform(const float* __restrict__ conv_w, unsigned short* __restrict__ Wt) {
  int idx = blockIdx.x * blockDim.x + threadIdx.x;
  if (idx >= COUT * KSTR) return;
  int o = idx / KSTR;
  int k = idx - o * KSTR;
  unsigned short v = 0;
  if (k < KTOT) {
    int c = k / 9, r = k - c * 9, ky = r / 3, kx = r - ky * 3;
    v = f32_bf16(conv_w[((o * CIN + c) * 3 + ky) * 3 + kx]);
  }
  Wt[idx] = v;
}

// ---- K1: implicit-GEMM conv (bf16 WMMA, f32 acc), M=channels, N=pixels
__global__ __launch_bounds__(256) void conv_wmma(
    const float* __restrict__ zd, const float* __restrict__ yiq,
    const unsigned short* __restrict__ Wt, const float* __restrict__ bias,
    float* __restrict__ u0) {
  __shared__ float          R_sh[CIN * 3 * RW];   // 27.7 KB raw f32 patch [c][ky][col]
  __shared__ unsigned short A_sh[PIX * KSTR];     // 42.0 KB bf16 im2col tile [pixel][K]
  __shared__ unsigned short W_sh[COUT * KSTR];    // 42.0 KB bf16 weights   [chan][K]

  int wg = blockIdx.x;
  int x0 = (wg & 3) * PIX;
  int y  = (wg >> 2) & 255;
  int n  = wg >> 10;
  int tid  = threadIdx.x;
  int lane = tid & 31;
  int wave = tid >> 5;

  // --- TDM: DMA the whole packed weight block (64*328 bf16, flat 1-D tile) into LDS,
  // overlapped with the async input staging below. Issued once (wave 0; EXEC ignored).
#if HAVE_TDM
  if (wave == 0) {
    unsigned long long ga = (unsigned long long)(size_t)Wt;
    const unsigned nelem = COUT * KSTR;              // 20992 2-byte elements
    tdm_g0_t g0;
    g0.x = 1u;                                       // count=1, load, not restore
    g0.y = lds_off(&W_sh[0]);                        // LDS byte address
    g0.z = (unsigned)ga;                             // global_addr[31:0]
    g0.w = ((unsigned)(ga >> 32) & 0x01FFFFFFu) | 0x80000000u;  // addr[56:32] | type=2
    tdm_g1_t g1;
    g1[0] = 0x00010000;                              // workgroup_mask=0, data_size=1 (2B)
    g1[1] = (int)((nelem & 0xFFFFu) << 16);          // tensor_dim0[15:0]
    g1[2] = (int)((nelem >> 16) | (1u << 16));       // tensor_dim0[31:16] | tensor_dim1=1
    g1[3] = (int)((nelem & 0xFFFFu) << 16);          // tile_dim0 = nelem
    g1[4] = 0;                                       // tile_dim1=0 (1-D), tile_dim2=0
    g1[5] = (int)nelem;                              // tensor_dim0_stride[31:0]
    g1[6] = 0; g1[7] = 0;
    tdm_g24_t gz = {0, 0, 0, 0};
#if __clang_major__ >= 23
    tdm_g1_t gz8 = {0, 0, 0, 0, 0, 0, 0, 0};
    __builtin_amdgcn_tensor_load_to_lds(g0, g1, gz, gz, gz8, 0);
#else
    __builtin_amdgcn_tensor_load_to_lds(g0, g1, gz, gz, 0);
#endif
  }
#else
  for (int e = tid; e < (COUT * KSTR) / 2; e += 256)
    async_ld_b32(lds_off((const char*)W_sh + e * 4), (const char*)Wt + e * 4);
#endif

  // --- Phase 1: async-stage raw input patch (each element fetched once per WG).
  // col 0 <-> xx = x0-1 ... col 65 <-> xx = x0+64. OOB zeroed with disjoint ds stores.
  for (int e = tid; e < CIN * 3 * 66; e += 256) {
    int c   = e / 198;
    int rem = e - c * 198;
    int row = rem / 66;
    int col = rem - row * 66;
    int yy = y + row - 1;
    int xx = x0 + col - 1;
    unsigned dst = lds_off(&R_sh[(c * 3 + row) * RW + col]);
    if (yy >= 0 && yy < IH && xx >= 0 && xx < IW) {
      const float* g = (c < 31) ? &zd[((n * 31 + c) * IH + yy) * IW + xx]
                                : &yiq[((n * 3 + (c - 31)) * IH + yy) * IW + xx];
      async_ld_b32(dst, g);
    } else {
      R_sh[(c * 3 + row) * RW + col] = 0.f;
    }
  }
  wait_async0();
  __syncthreads();

  // --- Phase 2: bf16 im2col expansion from LDS (9x broadcast stays on-chip)
  for (int e = tid; e < PIX * KSTR; e += 256) {
    int p = e & 63;           // pixel
    int k = e >> 6;           // K index 0..327
    unsigned short v = 0;
    if (k < KTOT) {
      int c = k / 9, r = k - c * 9, ky = r / 3, kx = r - ky * 3;
      v = f32_bf16(R_sh[(c * 3 + ky) * RW + (p + kx)]);
    }
    A_sh[p * KSTR + k] = v;
  }
#if HAVE_TDM
  if (wave == 0) __builtin_amdgcn_s_wait_tensorcnt(0);
#endif
  __syncthreads();

  int pg = wave & 3;        // 4 pixel groups of 16 (GEMM N)
  int cg = wave >> 5 ? 0 : (wave >> 2);  // channel half (0/1)
  cg = wave >> 2;
  int ln = lane & 15;
  int hi = lane >> 4;
  int pbase = (pg * 16 + ln) * KSTR;    // B: im2col row of this lane's pixel

  v8f acc0 = {}; v8f acc1 = {};
#pragma unroll
  for (int kk = 0; kk < KPAD; kk += 32) {
    // B fragment (K x 16 pixels): v0-3 = 8 elems at kk+8*hi, v4-7 at kk+16+8*hi
    union { uint4 q[2]; v16bf v; } B;
    B.q[0] = *(const uint4*)&A_sh[pbase + kk + 8 * hi];
    B.q[1] = *(const uint4*)&A_sh[pbase + kk + 16 + 8 * hi];
#pragma unroll
    for (int og = 0; og < 2; ++og) {
      int o = cg * 32 + og * 16 + ln;   // A row = channel
      // A fragment (16 chan x 32 K): v0-3 = 8 elems at kk+8*hi, v4-7 at kk+16+8*hi
      union { uint4 q[2]; v16bf v; } A;
      A.q[0] = *(const uint4*)&W_sh[o * KSTR + kk + 8 * hi];
      A.q[1] = *(const uint4*)&W_sh[o * KSTR + kk + 16 + 8 * hi];
      if (og == 0)
        acc0 = __builtin_amdgcn_wmma_f32_16x16x32_bf16(false, A.v, false, B.v,
                                                       (short)0, acc0, false, false);
      else
        acc1 = __builtin_amdgcn_wmma_f32_16x16x32_bf16(false, A.v, false, B.v,
                                                       (short)0, acc1, false, false);
    }
  }

  // epilogue: lane = pixel col (coalesced x), VGPR r -> channel row (M = r + 8*hi)
#pragma unroll
  for (int og = 0; og < 2; ++og) {
    v8f a = og ? acc1 : acc0;
#pragma unroll
    for (int r = 0; r < 8; ++r) {
      int o = cg * 32 + og * 16 + r + 8 * hi;
      int x = x0 + pg * 16 + ln;
      float val = a[r] + bias[o];
      val = val >= 0.f ? val : 0.2f * val;
      u0[(((size_t)n * COUT + o) * IH + y) * IW + x] = val;
    }
  }
}

// ---- K2: all 10 TV iterations fused in LDS (halo-10 tiles); u - soft(u,t) == clamp(u,+-t)
__global__ __launch_bounds__(256) void tv_fused(
    const float* __restrict__ u0, const float* __restrict__ lmbd,
    float* __restrict__ out) {
  __shared__ float Xs[LOADT * LSTR];
  __shared__ float Ua[LOADT * LSTR];
  __shared__ float Ub[LOADT * LSTR];

  int plane = blockIdx.y;                      // n*64 + c
  int t  = blockIdx.x;                         // 36 tiles (6x6 of 48)
  int ty = (t / 6) * TVT - HALO;
  int tx = (t % 6) * TVT - HALO;
  int tid = threadIdx.x;
  const float* src = u0 + (size_t)plane * (IH * IW);
  float thr = 1.0f / lmbd[0];

  for (int e = tid; e < LOADT * LOADT; e += 256) {
    int li = e / LOADT, lj = e - li * LOADT;
    int gi = ty + li, gj = tx + lj;
    if (gi >= 0 && gi < IH && gj >= 0 && gj < IW)
      async_ld_b32(lds_off(&Xs[li * LSTR + lj]), &src[gi * IW + gj]);
    else
      Xs[li * LSTR + lj] = 0.f;
  }
  wait_async0();
  __syncthreads();

  for (int it = 0; it < 10; ++it) {
    const float* U = (it == 0) ? Xs : ((it & 1) ? Ua : Ub);
    float* D = (it & 1) ? Ub : Ua;
    for (int e = tid; e < LOADT * LOADT; e += 256) {
      int li = e / LOADT, lj = e - li * LOADT;
      int gi = ty + li, gj = tx + lj;
      float uc = U[li * LSTR + lj];
      float sxR = 0.f, sxL = 0.f, syD = 0.f, syU = 0.f;
      if (lj < LOADT - 1 && gj < IW - 1) {
        float g = U[li * LSTR + lj + 1] - uc;
        sxR = fminf(fmaxf(g, -thr), thr);
      }
      if (lj > 0 && gj > 0) {
        float g = uc - U[li * LSTR + lj - 1];
        sxL = fminf(fmaxf(g, -thr), thr);
      }
      if (li < LOADT - 1 && gi < IH - 1) {
        float g = U[(li + 1) * LSTR + lj] - uc;
        syD = fminf(fmaxf(g, -thr), thr);
      }
      if (li > 0 && gi > 0) {
        float g = uc - U[(li - 1) * LSTR + lj];
        syU = fminf(fmaxf(g, -thr), thr);
      }
      float r  = (sxL - sxR) + (syU - syD);
      float xv = Xs[li * LSTR + lj];
      D[li * LSTR + lj] = uc - TAU * ((uc - xv) + r);
    }
    __syncthreads();
  }

  for (int e = tid; e < TVT * TVT; e += 256) {   // final result in Ub (it=9 -> Ub)
    int li = HALO + e / TVT, lj = HALO + e % TVT;
    int gi = ty + li, gj = tx + lj;
    if (gi < IH && gj < IW)
      out[(size_t)plane * (IH * IW) + gi * IW + gj] = Ub[li * LSTR + lj];
  }
}

// ---- K3: per-(n,c)-plane partial sums (deterministic tree reduce, b128 loads)
__global__ __launch_bounds__(256) void bn_reduce(const float* __restrict__ y,
                                                 float* __restrict__ part) {
  __shared__ float s1[256], s2[256];
  int plane = blockIdx.x;
  const float* p = y + (size_t)plane * (IH * IW);
  float a = 0.f, b = 0.f;
  for (int i = threadIdx.x * 4; i < IH * IW; i += 1024) {
    float4 v = *(const float4*)(p + i);
    a += v.x + v.y + v.z + v.w;
    b += v.x * v.x + v.y * v.y + v.z * v.z + v.w * v.w;
  }
  s1[threadIdx.x] = a; s2[threadIdx.x] = b;
  __syncthreads();
  for (int off = 128; off > 0; off >>= 1) {
    if (threadIdx.x < off) {
      s1[threadIdx.x] += s1[threadIdx.x + off];
      s2[threadIdx.x] += s2[threadIdx.x + off];
    }
    __syncthreads();
  }
  if (threadIdx.x == 0) { part[plane] = s1[0]; part[512 + plane] = s2[0]; }
}

// ---- K4: combine partials (fixed order) -> mean / inv-std per channel
__global__ void bn_stats(const float* __restrict__ part, float* __restrict__ stats) {
  int c = threadIdx.x;
  if (c < COUT) {
    float s = 0.f, q = 0.f;
    for (int n = 0; n < NIMG; ++n) {
      s += part[n * COUT + c];
      q += part[512 + n * COUT + c];
    }
    const float M = (float)NIMG * IH * IW;
    float mean = s / M;
    float var  = q / M - mean * mean;
    stats[c] = mean;
    stats[COUT + c] = rsqrtf(var + BN_EPS);
  }
}

// ---- K5: normalize in place (b128 load/store)
__global__ __launch_bounds__(256) void bn_norm(float* __restrict__ y,
    const float* __restrict__ stats, const float* __restrict__ gamma,
    const float* __restrict__ beta) {
  size_t base = ((size_t)blockIdx.x * 256 + threadIdx.x) * 4;
  int c = (int)((base >> 16) & 63);
  float mean = stats[c], rs = stats[COUT + c], g = gamma[c], bt = beta[c];
  float4 v = *(const float4*)(y + base);
  v.x = (v.x - mean) * rs * g + bt;
  v.y = (v.y - mean) * rs * g + bt;
  v.z = (v.z - mean) * rs * g + bt;
  v.w = (v.w - mean) * rs * g + bt;
  *(float4*)(y + base) = v;
}

extern "C" void kernel_launch(void* const* d_in, const int* in_sizes, int n_in,
                              void* d_out, int out_size, void* d_ws, size_t ws_size,
                              hipStream_t stream) {
  const float* zd     = (const float*)d_in[0];
  const float* yiq    = (const float*)d_in[1];
  const float* conv_w = (const float*)d_in[2];
  const float* conv_b = (const float*)d_in[3];
  const float* gamma  = (const float*)d_in[4];
  const float* beta   = (const float*)d_in[5];
  const float* lmbd   = (const float*)d_in[6];
  float* out = (float*)d_out;

  char* ws = (char*)d_ws;
  float*          u0    = (float*)ws;                              // 134217728 B
  unsigned short* Wt    = (unsigned short*)(ws + 134217728);       // 41984 B
  float*          part  = (float*)(ws + 134217728 + 65536);        // 1024 floats
  float*          stats = part + 1024;                             // 128 floats

  wt_xform <<<(COUT * KSTR + 255) / 256, 256, 0, stream>>>(conv_w, Wt);
  conv_wmma<<<NIMG * IH * (IW / PIX), 256, 0, stream>>>(zd, yiq, Wt, conv_b, u0);
  tv_fused <<<dim3(36, NIMG * COUT), 256, 0, stream>>>(u0, lmbd, out);
  bn_reduce<<<NIMG * COUT, 256, 0, stream>>>(out, part);
  bn_stats <<<1, 64, 0, stream>>>(part, stats);
  bn_norm  <<<(NIMG * COUT * IH * IW) / 1024, 256, 0, stream>>>(out, stats, gamma, beta);
}
